// MDTA_33921651704024
// MI455X (gfx1250) — matrix operand order repaired and checked by reading, not compiled, not run on hardware
//
#include <hip/hip_runtime.h>

// ---------------------------------------------------------------------------
// MDTA (channel attention) for MI455X / gfx1250, wave32 + WMMA f16.
//
// Shapes: B=4, H=W=256 -> t=65536 tokens per batch, C=128, 4 heads of s=32.
//
// Pass 1 (k1): per 512-token tile: LN -> xn(f16, LDS) -> Q,K = xn@W (WMMA)
//              staged transposed in LDS -> per-head K^T Q 32x32 score
//              partials accumulated in WMMA accumulators across 4 chunks,
//              written to workspace (one slot per workgroup: deterministic).
// Pass 2 (k2): reduce partials, /alpha, row softmax -> attn (ws).
// Pass 3 (k3): per 128-token tile: LN -> V = xn@Wv (WMMA) -> O = V@attn
//              (WMMA) -> y = O@Wf + bf + xn (WMMA), store.
//
// All matmuls use v_wmma_f32_16x16x32_f16 with the CDNA5 wave32 fragment
// layouts (A: per-lane row, K split 0-7/16-23 vs 8-15/24-31; B: lane=N,
// same K split as A; C/D: 8 VGPRs,
// M = r + 8*(lane>=16), N = lane%16).  LDS tiles use 16B-aligned padded
// strides so b128 LDS ops are legal.
// ---------------------------------------------------------------------------

#define C_DIM 128
#define T_TOT 65536
#define LSTR  136   // f16 tile row stride (halfs): 272B, multiple of 16B
#define FSTR  132   // f32 tile row stride (floats): 528B, multiple of 16B
#define EPS_LN 1e-5f

typedef __attribute__((ext_vector_type(2)))  _Float16 v2h;
typedef __attribute__((ext_vector_type(4)))  _Float16 v4h;
typedef __attribute__((ext_vector_type(8)))  _Float16 v8h;
typedef __attribute__((ext_vector_type(16))) _Float16 v16h;
typedef __attribute__((ext_vector_type(8)))  float    v8f;

__device__ __forceinline__ v8f wmma_f16(v16h a, v16h b, v8f c) {
  // (neg_a, A, neg_b, B, c_mod, C, reuse_a, reuse_b)
  return __builtin_amdgcn_wmma_f32_16x16x32_f16(false, a, false, b,
                                                (short)0, c, false, false);
}

// A fragment (M=16,K=32) from a row-major [M][K] f16 LDS tile.
// lane<16: halfs 0-7 -> K+0..7, 8-15 -> K+16..23 ; lane>=16: +8 / +24.
__device__ __forceinline__ v16h frag_A(const _Float16* base, int mrow,
                                       int kbase, int stride) {
  const int lane = threadIdx.x & 31;
  const int hi = lane >> 4;
  const _Float16* p = base + (mrow + (lane & 15)) * stride + kbase + hi * 8;
  v8h lo = *(const v8h*)p;          // 16B aligned
  v8h hv = *(const v8h*)(p + 16);   // 16B aligned
  return __builtin_shufflevector(lo, hv, 0, 1, 2, 3, 4, 5, 6, 7,
                                 8, 9, 10, 11, 12, 13, 14, 15);
}

// B fragment (K=32,N=16) from an [N][K] f16 LDS tile (B stored transposed).
// lane<16: halfs 0-7 -> K+0..7, 8-15 -> K+16..23 ; lane>=16: +8 / +24.
__device__ __forceinline__ v16h frag_B(const _Float16* base, int ncol,
                                       int kbase, int stride) {
  const int lane = threadIdx.x & 31;
  const int hi = lane >> 4;
  const _Float16* p = base + (ncol + (lane & 15)) * stride + kbase + hi * 8;
  v8h lo = *(const v8h*)p;
  v8h hv = *(const v8h*)(p + 16);
  return __builtin_shufflevector(lo, hv, 0, 1, 2, 3, 4, 5, 6, 7,
                                 8, 9, 10, 11, 12, 13, 14, 15);
}

// Scatter a 16x16 f32 D tile (+bias) into an [N][T] transposed f16 LDS tile.
// D layout: M = r + 8*(lane>=16), N = lane%16.  Pairs (r, r+1) are contiguous
// in the T dimension -> packed v2h stores.
__device__ __forceinline__ void scatter_D_nt(_Float16* base, int nbase,
                                             int tbase, int stride, v8f d,
                                             float bias) {
  const int lane = threadIdx.x & 31;
  const int hi = lane >> 4;
  _Float16* p = base + (nbase + (lane & 15)) * stride + tbase + hi * 8;
#pragma unroll
  for (int r = 0; r < 8; r += 2) {
    v2h w = { (_Float16)(d[r] + bias), (_Float16)(d[r + 1] + bias) };
    *(v2h*)(p + r) = w;
  }
}

// Scatter a 16x16 f32 D tile (+bias) into a token-major [T][C] f16 LDS tile.
__device__ __forceinline__ void scatter_D_tc(_Float16* base, int tbase,
                                             int nbase, int stride, v8f d,
                                             float bias) {
  const int lane = threadIdx.x & 31;
  const int hi = lane >> 4;
  const int n = nbase + (lane & 15);
#pragma unroll
  for (int r = 0; r < 8; ++r)
    base[(tbase + r + hi * 8) * stride + n] = (_Float16)(d[r] + bias);
}

// LayerNorm a 128-token chunk: one wave per token, 4 channels per lane.
// Writes f16 tile (and optionally f32 tile for the residual path).
template <bool WF>
__device__ __forceinline__ void ln_stage(const float* __restrict__ xg,
                                         const float* __restrict__ gamma,
                                         const float* __restrict__ beta,
                                         _Float16* sXn, float* sXnF) {
  const int wave = threadIdx.x >> 5;
  const int lane = threadIdx.x & 31;
  const float4 g  = ((const float4*)gamma)[lane];
  const float4 be = ((const float4*)beta)[lane];
#pragma unroll 4
  for (int it = 0; it < 16; ++it) {
    const int t = it * 8 + wave;
    const float4 v = ((const float4*)(xg + (size_t)t * C_DIM))[lane];
    float s = v.x + v.y + v.z + v.w;
#pragma unroll
    for (int o = 16; o > 0; o >>= 1) s += __shfl_xor(s, o, 32);
    const float mu = s * (1.0f / C_DIM);
    const float dx = v.x - mu, dy = v.y - mu, dz = v.z - mu, dw = v.w - mu;
    float q = dx * dx + dy * dy + dz * dz + dw * dw;
#pragma unroll
    for (int o = 16; o > 0; o >>= 1) q += __shfl_xor(q, o, 32);
    const float rs = rsqrtf(q * (1.0f / C_DIM) + EPS_LN);
    const float nx = dx * rs * g.x + be.x, ny = dy * rs * g.y + be.y;
    const float nz = dz * rs * g.z + be.z, nw = dw * rs * g.w + be.w;
    v4h hv = { (_Float16)nx, (_Float16)ny, (_Float16)nz, (_Float16)nw };
    *(v4h*)(sXn + t * LSTR + lane * 4) = hv;
    if constexpr (WF)
      *(float4*)(sXnF + t * FSTR + lane * 4) = make_float4(nx, ny, nz, nw);
  }
}

// W (row-major [K=Cin][N=Cout] f32 in global) -> [N][K] f16 LDS tile.
__device__ __forceinline__ void load_weight_T(_Float16* dst,
                                              const float* __restrict__ W) {
#pragma unroll 4
  for (int i = threadIdx.x; i < C_DIM * C_DIM; i += 256) {
    const int k = i >> 7, n = i & 127;     // coalesced read over n
    dst[n * LSTR + k] = (_Float16)W[i];
  }
}

// ---------------------------------------------------------------------------
// Kernel 1: per-head score partials.  512 workgroups (4 b x 128 tiles of 512
// tokens, processed in 4 chunks of 128).
// ---------------------------------------------------------------------------
__global__ __launch_bounds__(256) void k1_scores(
    const float* __restrict__ x, const float* __restrict__ gamma,
    const float* __restrict__ beta, const float* __restrict__ Wq,
    const float* __restrict__ bq, const float* __restrict__ Wk,
    const float* __restrict__ bk, float* __restrict__ partials) {
  __shared__ __align__(16) _Float16 sXn[C_DIM * LSTR];  // [token][cin]
  __shared__ __align__(16) _Float16 sQt[C_DIM * LSTR];  // [cout][token]
  __shared__ __align__(16) _Float16 sKt[C_DIM * LSTR];  // [cout][token]
  __shared__ __align__(16) _Float16 sWq[C_DIM * LSTR];  // [cout][cin]
  __shared__ __align__(16) _Float16 sWk[C_DIM * LSTR];

  const int wg = blockIdx.x;
  const int b = wg >> 7;
  const size_t tok0 = (size_t)b * T_TOT + (size_t)(wg & 127) * 512;

  load_weight_T(sWq, Wq);
  load_weight_T(sWk, Wk);

  const int wave = threadIdx.x >> 5;
  const int lane = threadIdx.x & 31;
  const int h = wave >> 1;        // head for the score stage
  const int jt = wave & 1;        // j half-tile for the score stage
  const int mt = wave * 16;       // token row block for the projection stage

  v8f accS0 = {};                 // scores i = 0..15
  v8f accS1 = {};                 // scores i = 16..31

  for (int ch = 0; ch < 4; ++ch) {
    const float* xg = x + (tok0 + (size_t)ch * 128) * C_DIM;
    if (ch + 1 < 4) {             // pull next 64KB chunk toward L2/WGP$
      const float* nx = x + (tok0 + (size_t)(ch + 1) * 128) * C_DIM;
      __builtin_prefetch(nx + threadIdx.x * 64, 0, 0);
    }
    __syncthreads();              // sQt/sKt readers done; weights ready (ch=0)
    ln_stage<false>(xg, gamma, beta, sXn, nullptr);
    __syncthreads();

    // Q,K projections: wave -> 16-token row block, loop over cout blocks.
    {
      const v16h a0 = frag_A(sXn, mt, 0, LSTR);
      const v16h a1 = frag_A(sXn, mt, 32, LSTR);
      const v16h a2 = frag_A(sXn, mt, 64, LSTR);
      const v16h a3 = frag_A(sXn, mt, 96, LSTR);
#pragma unroll 2
      for (int nt = 0; nt < 8; ++nt) {
        const float biq = bq[nt * 16 + (lane & 15)];
        const float bik = bk[nt * 16 + (lane & 15)];
        v8f aq = {};
        aq = wmma_f16(a0, frag_B(sWq, nt * 16, 0, LSTR), aq);
        aq = wmma_f16(a1, frag_B(sWq, nt * 16, 32, LSTR), aq);
        aq = wmma_f16(a2, frag_B(sWq, nt * 16, 64, LSTR), aq);
        aq = wmma_f16(a3, frag_B(sWq, nt * 16, 96, LSTR), aq);
        scatter_D_nt(sQt, nt * 16, mt, LSTR, aq, biq);
        v8f ak = {};
        ak = wmma_f16(a0, frag_B(sWk, nt * 16, 0, LSTR), ak);
        ak = wmma_f16(a1, frag_B(sWk, nt * 16, 32, LSTR), ak);
        ak = wmma_f16(a2, frag_B(sWk, nt * 16, 64, LSTR), ak);
        ak = wmma_f16(a3, frag_B(sWk, nt * 16, 96, LSTR), ak);
        scatter_D_nt(sKt, nt * 16, mt, LSTR, ak, bik);
      }
    }
    __syncthreads();

    // S_h[i][j] += sum_t K[t][i] * Q[t][j]: A = K^T from sKt, B = Q from sQt.
#pragma unroll
    for (int ks = 0; ks < 4; ++ks) {
      const v16h bfr = frag_B(sQt, h * 32 + jt * 16, ks * 32, LSTR);
      accS0 = wmma_f16(frag_A(sKt, h * 32, ks * 32, LSTR), bfr, accS0);
      accS1 = wmma_f16(frag_A(sKt, h * 32 + 16, ks * 32, LSTR), bfr, accS1);
    }
  }

  // partials[wg][h][i][j]
  float* dst = partials + ((size_t)wg * 4 + h) * 1024;
  const int hi = lane >> 4, j = jt * 16 + (lane & 15);
#pragma unroll
  for (int r = 0; r < 8; ++r) {
    dst[(r + 8 * hi) * 32 + j] = accS0[r];
    dst[(16 + r + 8 * hi) * 32 + j] = accS1[r];
  }
}

// ---------------------------------------------------------------------------
// Kernel 2: reduce 128 partials per (b,h), /alpha, row softmax -> attn.
// Grid 16, block 1024 (wave = row i, lane = col j).
// ---------------------------------------------------------------------------
__global__ __launch_bounds__(1024) void k2_softmax(
    const float* __restrict__ partials, const float* __restrict__ alpha,
    float* __restrict__ attn) {
  const int bh = blockIdx.x;
  const int b = bh >> 2, h = bh & 3;
  const int i = threadIdx.x >> 5, j = threadIdx.x & 31;
  float s = 0.0f;
  for (int t = 0; t < 128; ++t)
    s += partials[(((size_t)(b * 128 + t) * 4 + h) << 10) + i * 32 + j];
  s /= alpha[0];
  float m = s;
#pragma unroll
  for (int o = 16; o > 0; o >>= 1) m = fmaxf(m, __shfl_xor(m, o, 32));
  const float e = __expf(s - m);
  float d = e;
#pragma unroll
  for (int o = 16; o > 0; o >>= 1) d += __shfl_xor(d, o, 32);
  attn[((size_t)bh << 10) + i * 32 + j] = e / d;
}

// ---------------------------------------------------------------------------
// Kernel 3: V = xn@Wv, O = V@attn, y = O@Wf + bf + xn.  2048 workgroups of
// 128 tokens each.
// ---------------------------------------------------------------------------
__global__ __launch_bounds__(256) void k3_out(
    const float* __restrict__ x, const float* __restrict__ gamma,
    const float* __restrict__ beta, const float* __restrict__ Wv,
    const float* __restrict__ bv, const float* __restrict__ Wf,
    const float* __restrict__ bf, const float* __restrict__ attnG,
    float* __restrict__ y) {
  __shared__ __align__(16) _Float16 sXn[C_DIM * LSTR];   // [token][cin] f16
  __shared__ __align__(16) float    sXnF[C_DIM * FSTR];  // residual f32
  __shared__ __align__(16) _Float16 sV[C_DIM * LSTR];    // [token][c] f16
  __shared__ __align__(16) _Float16 sO[C_DIM * LSTR];    // [token][c] f16
  __shared__ __align__(16) _Float16 sWv[C_DIM * LSTR];
  __shared__ __align__(16) _Float16 sWf[C_DIM * LSTR];
  __shared__ __align__(16) _Float16 sAt[4 * 32 * 32];    // [h][j][i] f16

  const int wg = blockIdx.x;
  const int b = wg >> 9;
  const size_t tok0 = (size_t)b * T_TOT + (size_t)(wg & 511) * 128;

  load_weight_T(sWv, Wv);
  load_weight_T(sWf, Wf);
  for (int idx = threadIdx.x; idx < 4096; idx += 256) {   // attn -> [h][j][i]
    const int hh = idx >> 10, r = idx & 1023, ii = r >> 5, jj = r & 31;
    sAt[(hh * 32 + jj) * 32 + ii] =
        (_Float16)attnG[((size_t)(b * 4 + hh) << 10) + r];
  }
  ln_stage<true>(x + tok0 * C_DIM, gamma, beta, sXn, sXnF);
  __syncthreads();

  const int wave = threadIdx.x >> 5;
  const int lane = threadIdx.x & 31;
  const int mt = wave * 16;

  // V projection.
  {
    const v16h a0 = frag_A(sXn, mt, 0, LSTR);
    const v16h a1 = frag_A(sXn, mt, 32, LSTR);
    const v16h a2 = frag_A(sXn, mt, 64, LSTR);
    const v16h a3 = frag_A(sXn, mt, 96, LSTR);
#pragma unroll 2
    for (int nt = 0; nt < 8; ++nt) {
      v8f acc = {};
      acc = wmma_f16(a0, frag_B(sWv, nt * 16, 0, LSTR), acc);
      acc = wmma_f16(a1, frag_B(sWv, nt * 16, 32, LSTR), acc);
      acc = wmma_f16(a2, frag_B(sWv, nt * 16, 64, LSTR), acc);
      acc = wmma_f16(a3, frag_B(sWv, nt * 16, 96, LSTR), acc);
      scatter_D_tc(sV, mt, nt * 16, LSTR, acc, bv[nt * 16 + (lane & 15)]);
    }
  }
  __syncthreads();

  // O[t][h*32+j] = sum_i V[t][h*32+i] * attn[h][i][j].
  {
    const int h = wave >> 1, jt = wave & 1;
    const v16h bfr = frag_B(sAt + h * 1024, jt * 16, 0, 32);
#pragma unroll 2
    for (int m = 0; m < 8; ++m) {
      v8f acc = {};
      acc = wmma_f16(frag_A(sV, m * 16, h * 32, LSTR), bfr, acc);
      scatter_D_tc(sO, m * 16, h * 32 + jt * 16, LSTR, acc, 0.0f);
    }
  }
  __syncthreads();

  // y = O @ Wf + bf + xn.
  {
    const v16h a0 = frag_A(sO, mt, 0, LSTR);
    const v16h a1 = frag_A(sO, mt, 32, LSTR);
    const v16h a2 = frag_A(sO, mt, 64, LSTR);
    const v16h a3 = frag_A(sO, mt, 96, LSTR);
    float* yb = y + tok0 * C_DIM;
    const int hi = lane >> 4;
#pragma unroll 2
    for (int nt = 0; nt < 8; ++nt) {
      v8f acc = {};
      acc = wmma_f16(a0, frag_B(sWf, nt * 16, 0, LSTR), acc);
      acc = wmma_f16(a1, frag_B(sWf, nt * 16, 32, LSTR), acc);
      acc = wmma_f16(a2, frag_B(sWf, nt * 16, 64, LSTR), acc);
      acc = wmma_f16(a3, frag_B(sWf, nt * 16, 96, LSTR), acc);
      const int n = nt * 16 + (lane & 15);
      const float bb = bf[n];
#pragma unroll
      for (int r = 0; r < 8; ++r) {
        const int m = mt + r + 8 * hi;
        yb[(size_t)m * C_DIM + n] = acc[r] + bb + sXnF[m * FSTR + n];
      }
    }
  }
}

// ---------------------------------------------------------------------------
extern "C" void kernel_launch(void* const* d_in, const int* in_sizes, int n_in,
                              void* d_out, int out_size, void* d_ws,
                              size_t ws_size, hipStream_t stream) {
  const float* x     = (const float*)d_in[0];
  const float* gamma = (const float*)d_in[1];
  const float* beta  = (const float*)d_in[2];
  const float* Wq    = (const float*)d_in[3];
  const float* bq    = (const float*)d_in[4];
  const float* Wk    = (const float*)d_in[5];
  const float* bk    = (const float*)d_in[6];
  const float* Wv    = (const float*)d_in[7];
  const float* bv    = (const float*)d_in[8];
  const float* Wf    = (const float*)d_in[9];
  const float* bf    = (const float*)d_in[10];
  const float* alpha = (const float*)d_in[11];
  float* yout = (float*)d_out;

  float* partials = (float*)d_ws;                        // 512*4*1024 f32 = 8MB
  float* attn     = partials + (size_t)512 * 4 * 1024;   // 16*1024 f32

  k1_scores<<<512, 256, 0, stream>>>(x, gamma, beta, Wq, bq, Wk, bk, partials);
  k2_softmax<<<16, 1024, 0, stream>>>(partials, alpha, attn);
  k3_out<<<2048, 256, 0, stream>>>(x, gamma, beta, Wv, bv, Wf, bf, attn, yout);
}
